// KernelPointAggregation_25348896981217
// MI455X (gfx1250) — matrix operand order repaired and checked by reading, not compiled, not run on hardware
//
#include <hip/hip_runtime.h>
#include <hip/hip_bf16.h>

typedef __attribute__((ext_vector_type(16))) _Float16 v16h;
typedef __attribute__((ext_vector_type(8)))  float    v8f;

#define NN   20000
#define NEI  16
#define KK   4
#define DIM  64
#define WPB  5          // waves (nodes) per block; 20000 = 4000 * 5
#define WROW 72         // padded W row pitch in halves (bank-conflict-free)
#define MAXN 0.99999f   // (1-1e-5)/sqrt(c), c=1
#define KPE  0.66f

__device__ __forceinline__ float artanh_c(float v) {
    v = fminf(fmaxf(v, -1.f + 1e-7f), 1.f - 1e-7f);
    return 0.5f * logf((1.f + v) / (1.f - v));
}
// partner (other 32-dim half of same neighbor) reduction
__device__ __forceinline__ float psum(float v) { return v + __shfl_xor(v, 1, 32); }
// reduction over the 16 neighbors (same parity lanes)
__device__ __forceinline__ float nsum(float v) {
    v += __shfl_xor(v, 2, 32);
    v += __shfl_xor(v, 4, 32);
    v += __shfl_xor(v, 8, 32);
    v += __shfl_xor(v, 16, 32);
    return v;
}

// ---------------- init: W -> f16, expmap0 of kernel tangents and bias ----------
__global__ void kpa_init(const float* __restrict__ W, const float* __restrict__ b,
                         const float* __restrict__ kt,
                         _Float16* __restrict__ Wh, float* __restrict__ xk,
                         float* __restrict__ eb, float* __restrict__ xk2,
                         float* __restrict__ eb2) {
    int tid = threadIdx.x;
    for (int i = tid; i < KK * DIM * DIM; i += blockDim.x)
        Wh[i] = (_Float16)W[i];
    if (tid < KK) {
        float s2 = 0.f;
        for (int d = 0; d < DIM; ++d) { float v = kt[tid * DIM + d]; s2 += v * v; }
        float n = sqrtf(fmaxf(s2, 1e-14f));
        float sc = tanhf(n) / n;
        for (int d = 0; d < DIM; ++d) xk[tid * DIM + d] = kt[tid * DIM + d] * sc;
        xk2[tid] = sc * sc * s2;

        s2 = 0.f;
        for (int d = 0; d < DIM; ++d) { float v = b[tid * DIM + d]; s2 += v * v; }
        n = sqrtf(fmaxf(s2, 1e-14f));
        sc = tanhf(n) / n;
        for (int d = 0; d < DIM; ++d) eb[tid * DIM + d] = b[tid * DIM + d] * sc;
        eb2[tid] = sc * sc * s2;
    }
}

// per-wave LDS scratch: {u + xr} is dead before z is first written (same-wave
// DS ops are in-order on CDNA5), so they can share storage.
struct PreT { _Float16 u[16][DIM]; float xr[DIM]; };
union WaveScratch { PreT pre; float z[16][DIM]; };

// ---------------- main: one wave32 per node -----------------------------------
__global__ __launch_bounds__(WPB * 32) void
kpa_main(const float* __restrict__ x, const int* __restrict__ nei,
         const float* __restrict__ nmask, const _Float16* __restrict__ Wh,
         const float* __restrict__ xk, const float* __restrict__ eb,
         const float* __restrict__ xk2, const float* __restrict__ eb2,
         float* __restrict__ out) {
    __shared__ __align__(16) WaveScratch s_wv[WPB];          // 5 * 4KB
    __shared__ __align__(16) _Float16 s_W[KK * DIM * WROW];  // 36KB, padded rows
    __shared__ float s_xk[KK][DIM];
    __shared__ float s_eb[KK][DIM];
    __shared__ float s_xk2[KK], s_eb2[KK];

    const int tid  = threadIdx.x;
    const int wave = tid >> 5;
    const int lane = tid & 31;
    const int b    = blockIdx.x * WPB + wave;

    // warm L2 / WGP$ for the W copy (gfx1250 global_prefetch_b8)
    __builtin_prefetch((const void*)(Wh + (tid & 127) * 128), 0, 1);

    // block-cooperative: stage f16 W into LDS with 72-half row pitch
    for (int c = tid; c < KK * DIM * 8; c += blockDim.x) {
        int r = c >> 3, cc = c & 7;
        *(uint4*)&s_W[r * WROW + cc * 8] = *(const uint4*)&Wh[r * DIM + cc * 8];
    }
    // block-cooperative constant loads
    for (int i = tid; i < KK * DIM; i += blockDim.x) {
        s_xk[i / DIM][i % DIM] = xk[i];
        s_eb[i / DIM][i % DIM] = eb[i];
    }
    if (tid < KK) { s_xk2[tid] = xk2[tid]; s_eb2[tid] = eb2[tid]; }
    // per-wave center point
    s_wv[wave].pre.xr[lane]      = x[b * DIM + lane];
    s_wv[wave].pre.xr[lane + 32] = x[b * DIM + lane + 32];
    __syncthreads();

    // |x|^2 (replicated over wave)
    float xp = s_wv[wave].pre.xr[lane] * s_wv[wave].pre.xr[lane] +
               s_wv[wave].pre.xr[lane + 32] * s_wv[wave].pre.xr[lane + 32];
    const float x2 = nsum(psum(xp));

    // ---- neighbor phase: 2 lanes per neighbor, 32 dims per lane ----
    const int n  = lane >> 1;
    const int h  = lane & 1;
    const int db = 32 * h;

    const int   idx = nei[b * NEI + n];
    const float msk = nmask[b * NEI + n];

    float vv[32];
    const float4* yp = (const float4*)(x + idx * DIM + db);
#pragma unroll
    for (int j4 = 0; j4 < 8; ++j4) {
        float4 f = yp[j4];
        vv[4 * j4 + 0] = f.x; vv[4 * j4 + 1] = f.y;
        vv[4 * j4 + 2] = f.z; vv[4 * j4 + 3] = f.w;
    }
    float xyp = 0.f, y2p = 0.f;
#pragma unroll
    for (int j = 0; j < 32; ++j) {
        float xv = s_wv[wave].pre.xr[db + j];
        xyp += xv * vv[j];
        y2p += vv[j] * vv[j];
    }
    float xy = psum(xyp), y2 = psum(y2p);

    // m = mobius_add(-x, y); x0 = proj(m)  [exact collapse of transp/log/exp chain]
    float ca = 1.f - 2.f * xy + y2;
    float cb = 1.f - x2;
    float invden = 1.f / fmaxf(1.f - 2.f * xy + x2 * y2, 1e-15f);
    float m2p = 0.f;
#pragma unroll
    for (int j = 0; j < 32; ++j) {
        float md = (cb * vv[j] - ca * s_wv[wave].pre.xr[db + j]) * invden;
        vv[j] = md;
        m2p += md * md;
    }
    float m2 = psum(m2p);
    float nm = sqrtf(fmaxf(m2, 1e-14f));
    float pr = (nm > MAXN) ? (MAXN / nm) : 1.f;  // proj scale
    float q  = pr * pr * m2;                      // |x0|^2

    // u = logmap0(x0) -> LDS as f16 (A matrix, row = neighbor)
    float nq = sqrtf(fmaxf(q, 1e-14f));
    float us = artanh_c(nq) / nq * pr;
#pragma unroll
    for (int j = 0; j < 32; ++j)
        s_wv[wave].pre.u[n][db + j] = (_Float16)(us * vv[j]);

    // distances to kernel points from scalars only
    float wk[KK];
#pragma unroll
    for (int k = 0; k < KK; ++k) {
        float ppp = 0.f;
#pragma unroll
        for (int j = 0; j < 32; ++j) ppp += pr * vv[j] * s_xk[k][db + j];
        float p  = psum(ppp);
        float k2 = s_xk2[k];
        float aa = 1.f - 2.f * p + k2;
        float bb = 1.f - q;
        float num2 = aa * aa * q - 2.f * aa * bb * p + bb * bb * k2;
        float dn  = fmaxf(1.f - 2.f * p + q * k2, 1e-15f);
        float nmk = sqrtf(fmaxf(num2, 1e-14f)) / dn;
        float dis = 2.f * artanh_c(nmk);
        wk[k] = fmaxf(1.f - dis / KPE, 0.f) * msk;
    }

    // ---- matmul phase: D = u(16x64) * W_k^T via v_wmma_f32_16x16x32_f16 ----
    union UV { v16h v; uint4 q[2]; };
    const int col   = lane & 15;         // A row / B column / D column
    const int koffA = (lane >> 4) * 16;  // A: contiguous 16-half K blocks
    const int koffB = (lane >> 4) * 16;  // B: contiguous 16-half K blocks

    UV a0, a1;
    a0.q[0] = *(const uint4*)(&s_wv[wave].pre.u[col][ 0 + koffA]);
    a0.q[1] = *(const uint4*)(&s_wv[wave].pre.u[col][ 8 + koffA]);
    a1.q[0] = *(const uint4*)(&s_wv[wave].pre.u[col][32 + koffA]);
    a1.q[1] = *(const uint4*)(&s_wv[wave].pre.u[col][40 + koffA]);

    float outnum[32];
#pragma unroll
    for (int j = 0; j < 32; ++j) outnum[j] = 0.f;
    float outden = 0.f;

    for (int k = 0; k < KK; ++k) {
        // B operands now come from LDS (low latency, conflict-free pitch)
        v8f acc[4];
#pragma unroll
        for (int t = 0; t < 4; ++t) {
            const _Float16* wb = &s_W[(k * DIM + t * 16 + col) * WROW + koffB];
            UV b0, b1;
            b0.q[0] = *(const uint4*)(wb + 0);
            b0.q[1] = *(const uint4*)(wb + 8);
            b1.q[0] = *(const uint4*)(wb + 32);
            b1.q[1] = *(const uint4*)(wb + 40);
            v8f a = {0.f, 0.f, 0.f, 0.f, 0.f, 0.f, 0.f, 0.f};
            a = __builtin_amdgcn_wmma_f32_16x16x32_f16(false, a0.v, false, b0.v,
                                                       (short)0, a, false, false);
            a = __builtin_amdgcn_wmma_f32_16x16x32_f16(false, a1.v, false, b1.v,
                                                       (short)0, a, false, false);
            acc[t] = a;
        }
#pragma unroll
        for (int t = 0; t < 4; ++t) {
#pragma unroll
            for (int r = 0; r < 8; ++r)
                s_wv[wave].z[(lane >> 4) * 8 + r][t * 16 + col] = acc[t][r];
        }

        // ---- epilogue for kernel k (2 lanes per neighbor again) ----
        float z2p = 0.f, zep = 0.f;
#pragma unroll
        for (int j = 0; j < 32; ++j) {
            float zv = s_wv[wave].z[n][db + j];
            z2p += zv * zv;
            zep += zv * s_eb[k][db + j];
        }
        float z2 = psum(z2p), ze = psum(zep);
        float nz = sqrtf(fmaxf(z2, 1e-14f));
        float hs = tanhf(nz) / nz;                 // expmap0 scale
        float h2 = hs * hs * z2;                   // |h|^2
        float he = hs * ze;
        float e2 = s_eb2[k];
        float ca2 = 1.f + 2.f * he + e2;
        float cb2 = 1.f - h2;
        float idn = 1.f / fmaxf(1.f + 2.f * he + h2 * e2, 1e-15f);

        float mv[32];
        float mm2p = 0.f;
#pragma unroll
        for (int j = 0; j < 32; ++j) {
            float zv = s_wv[wave].z[n][db + j];
            float md = (ca2 * hs * zv + cb2 * s_eb[k][db + j]) * idn;
            mv[j] = md;
            mm2p += md * md;
        }
        float mm2 = psum(mm2p);
        float nmm = sqrtf(fmaxf(mm2, 1e-14f));
        float pr2 = (nmm > MAXN) ? (MAXN / nmm) : 1.f;
        float q2  = pr2 * pr2 * mm2;               // |proj(h+b)|^2
        float i1q = 1.f / (1.f + q2);
        float kx2 = 4.f * q2 * i1q * i1q;          // |p2k|^2
        float gam = rsqrtf(fmaxf(1.f - kx2, 1e-7f));
        float wgk = wk[k] * gam;
        float iwg = 1.f / fmaxf(nsum(wgk), 1e-10f);
        float fac = wgk * 2.f * pr2 * i1q;         // wg * p2k scale

        float mid2p = 0.f;
#pragma unroll
        for (int j = 0; j < 32; ++j) {
            float midd = nsum(fac * mv[j]) * iwg;  // weighted Klein mean
            mv[j] = midd;
            mid2p += midd * midd;
        }
        float mid2 = psum(mid2p);
        float g = rsqrtf(fmaxf(1.f - mid2, 1e-7f));
#pragma unroll
        for (int j = 0; j < 32; ++j) outnum[j] += g * mv[j];
        outden += g;
    }

    // ---- finalize: k2p -> proj -> relu(logmap0) -> expmap0 ----
    float iod = 1.f / fmaxf(outden, 1e-10f);
    float ov[32];
    float o2p = 0.f;
#pragma unroll
    for (int j = 0; j < 32; ++j) { float v = outnum[j] * iod; ov[j] = v; o2p += v * v; }
    float o2  = psum(o2p);
    float srt = sqrtf(fmaxf(1.f - o2, 1e-7f));
    float i1s = 1.f / (1.f + srt);                 // k2p scale
    float p2  = o2 * i1s * i1s;
    float npv = sqrtf(fmaxf(p2, 1e-14f));
    float pr3 = (npv > MAXN) ? (MAXN / npv) : 1.f; // proj scale
    float sc3 = i1s * pr3;
    float no2 = p2 * pr3 * pr3;
    float no  = sqrtf(fmaxf(no2, 1e-14f));
    float lsc = artanh_c(no) / no;                 // logmap0 scale
    float r2p = 0.f;
#pragma unroll
    for (int j = 0; j < 32; ++j) {
        float rv = fmaxf(lsc * ov[j] * sc3, 0.f);  // relu
        ov[j] = rv;
        r2p += rv * rv;
    }
    float r2  = psum(r2p);
    float nr  = sqrtf(fmaxf(r2, 1e-14f));
    float esc = tanhf(nr) / nr;                    // expmap0 scale
    if (n == 0) {
#pragma unroll
        for (int j = 0; j < 32; ++j) out[b * DIM + db + j] = esc * ov[j];
    }
}

extern "C" void kernel_launch(void* const* d_in, const int* in_sizes, int n_in,
                              void* d_out, int out_size, void* d_ws, size_t ws_size,
                              hipStream_t stream) {
    const float* x   = (const float*)d_in[0];
    const int*   nei = (const int*)d_in[1];
    const float* msk = (const float*)d_in[2];
    const float* W   = (const float*)d_in[3];
    const float* b   = (const float*)d_in[4];
    const float* kt  = (const float*)d_in[5];
    float* out = (float*)d_out;

    _Float16* Wh  = (_Float16*)d_ws;
    float*    xk  = (float*)((char*)d_ws + KK * DIM * DIM * sizeof(_Float16));
    float*    eb  = xk + KK * DIM;
    float*    xk2 = eb + KK * DIM;
    float*    eb2 = xk2 + KK;

    kpa_init<<<1, 256, 0, stream>>>(W, b, kt, Wh, xk, eb, xk2, eb2);
    kpa_main<<<NN / WPB, WPB * 32, 0, stream>>>(x, nei, msk, Wh, xk, eb, xk2, eb2, out);
}
